// AttentionBlock_78881369358732
// MI455X (gfx1250) — matrix operand order repaired and checked by reading, not compiled, not run on hardware
//
#include <hip/hip_runtime.h>

typedef __attribute__((ext_vector_type(16))) _Float16 v16h;
typedef __attribute__((ext_vector_type(8)))  float    v8f;
typedef _Float16 half_t;
typedef unsigned int u32;

#define Bdim 4
#define Ndim 2048
#define Cdim 768
#define Hdim 12
#define Ddim 64
#define Mdim (Bdim*Ndim)   // 8192

union FragH { v16h h; u32 u[8]; };

static __device__ inline v8f vzero() {
  v8f z = {0.f,0.f,0.f,0.f,0.f,0.f,0.f,0.f};
  return z;
}

// CDNA5 async global->LDS copy (ASYNCcnt-tracked), 16 bytes per lane.
static __device__ inline void async_b128(unsigned lds_off, const void* gaddr) {
  asm volatile("global_load_async_to_lds_b128 %0, %1, off"
               :: "v"(lds_off), "v"(gaddr) : "memory");
}
static __device__ inline void wait_async0() {
  asm volatile("s_wait_asynccnt 0x0" ::: "memory");
}

// ---------------------------------------------------------------------------
// Weight convert + transpose: dst[n*K + k] = (f16) src[k*Nc + n]
// ---------------------------------------------------------------------------
__global__ __launch_bounds__(256) void transpose_to_h(const float* __restrict__ src,
                                                      half_t* __restrict__ dst,
                                                      int K, int Nc) {
  int idx = blockIdx.x * 256 + threadIdx.x;
  if (idx >= K * Nc) return;
  int n = idx / K;
  int k = idx - n * K;
  dst[idx] = (half_t)src[(size_t)k * Nc + n];
}

// ---------------------------------------------------------------------------
// LayerNorm over C=768, one block per row, output f16
// ---------------------------------------------------------------------------
__global__ __launch_bounds__(256) void ln_kernel(const float* __restrict__ x,
                                                 const float* __restrict__ g,
                                                 const float* __restrict__ be,
                                                 half_t* __restrict__ out) {
  int row = blockIdx.x;
  const float* xr = x + (size_t)row * Cdim;
  float s = 0.f, s2 = 0.f;
  for (int i = threadIdx.x; i < Cdim; i += 256) {
    float v = xr[i]; s += v; s2 += v * v;
  }
  for (int m = 16; m >= 1; m >>= 1) {
    s  += __shfl_xor(s,  m, 32);
    s2 += __shfl_xor(s2, m, 32);
  }
  __shared__ float red[2][8];
  int wave = threadIdx.x >> 5, lane = threadIdx.x & 31;
  if (lane == 0) { red[0][wave] = s; red[1][wave] = s2; }
  __syncthreads();
  s = 0.f; s2 = 0.f;
  #pragma unroll
  for (int w = 0; w < 8; ++w) { s += red[0][w]; s2 += red[1][w]; }
  float mu  = s * (1.f / Cdim);
  float var = s2 * (1.f / Cdim) - mu * mu;
  float inv = rsqrtf(var + 1e-5f);
  for (int i = threadIdx.x; i < Cdim; i += 256)
    out[(size_t)row * Cdim + i] = (half_t)((xr[i] - mu) * inv * g[i] + be[i]);
}

// ---------------------------------------------------------------------------
// QKV GEMM: xn[8192,768] (f16) @ wqkvT[2304,768] (f16, BT layout)
// Block tile 256x128 (8 waves, 4x2), wave tile 64x64 (16 WMMA / k-step).
// Double-buffered async LDS staging: copies for step k+1 fly under step k.
// Writes Q,K as [B,H,N,D]; V as [B,H,D,N] (pre-transposed for attention).
// ---------------------------------------------------------------------------
__global__ __launch_bounds__(256) void gemm_qkv_kernel(
    const half_t* __restrict__ A, const half_t* __restrict__ BT,
    half_t* __restrict__ Qb, half_t* __restrict__ Kb, half_t* __restrict__ Vb) {
  __shared__ u32 sA[2][256][20];   // 16 data u32 + 4 pad (rows 80B, 16B-aligned)
  __shared__ u32 sB[2][128][20];

  const int tid = threadIdx.x;
  const int wave = tid >> 5, lane = tid & 31;
  const int hi = lane >> 4, l15 = lane & 15;
  const int kb4 = hi * 4, kb8 = hi * 8;
  const int wm = wave >> 1, wn = wave & 1;   // 4 x 2 waves
  const int mBase = blockIdx.x * 256;
  const int nBase = blockIdx.y * 128;
  const u32* Au = (const u32*)A;
  const u32* Bu = (const u32*)BT;

  auto stage = [&](int k0, int buf) {
    #pragma unroll
    for (int t = 0; t < 4; ++t) {
      int id = tid + t * 256;
      int r = id >> 2, c4 = (id & 3) * 4;
      async_b128((unsigned)(size_t)&sA[buf][r][c4],
                 Au + (size_t)(mBase + r) * (Cdim / 2) + (k0 >> 1) + c4);
    }
    #pragma unroll
    for (int t = 0; t < 2; ++t) {
      int id = tid + t * 256;
      int r = id >> 2, c4 = (id & 3) * 4;
      async_b128((unsigned)(size_t)&sB[buf][r][c4],
                 Bu + (size_t)(nBase + r) * (Cdim / 2) + (k0 >> 1) + c4);
    }
  };

  v8f acc[4][4];
  #pragma unroll
  for (int mt = 0; mt < 4; ++mt)
    #pragma unroll
    for (int nt = 0; nt < 4; ++nt) acc[mt][nt] = vzero();

  stage(0, 0);
  const int KSTEPS = Cdim / 32;   // 24
  for (int ks = 0; ks < KSTEPS; ++ks) {
    wait_async0();      // my copies for step ks landed
    __syncthreads();    // everyone's landed; prior reads of next buf done
    if (ks + 1 < KSTEPS) stage((ks + 1) * 32, (ks + 1) & 1);
    const int buf = ks & 1;

    FragH af[4], bf[4];
    #pragma unroll
    for (int t = 0; t < 4; ++t) {
      const u32* p = &sA[buf][wm * 64 + t * 16 + l15][0];
      #pragma unroll
      for (int i = 0; i < 4; ++i) { af[t].u[i] = p[kb4 + i]; af[t].u[4 + i] = p[8 + kb4 + i]; }
    }
    #pragma unroll
    for (int t = 0; t < 4; ++t) {
      const u32* p = &sB[buf][wn * 64 + t * 16 + l15][0];
      #pragma unroll
      for (int i = 0; i < 4; ++i) { bf[t].u[i] = p[kb4 + i]; bf[t].u[4 + i] = p[8 + kb4 + i]; }
    }
    #pragma unroll
    for (int mt = 0; mt < 4; ++mt)
      #pragma unroll
      for (int nt = 0; nt < 4; ++nt)
        acc[mt][nt] = __builtin_amdgcn_wmma_f32_16x16x32_f16(
            false, af[mt].h, false, bf[nt].h, (short)0, acc[mt][nt], false, false);
  }

  #pragma unroll
  for (int mt = 0; mt < 4; ++mt) {
    #pragma unroll
    for (int nt = 0; nt < 4; ++nt) {
      int col = nBase + wn * 64 + nt * 16 + l15;
      int sel = col / Cdim;
      int rem = col - sel * Cdim;
      int hh = rem >> 6, d = rem & 63;
      int m0 = mBase + wm * 64 + mt * 16 + hi * 8;
      if (sel == 2) {           // V: store transposed [B,H,D,N]
        #pragma unroll
        for (int r = 0; r < 8; ++r) {
          int m = m0 + r;
          int bb = m >> 11, ntok = m & (Ndim - 1);
          Vb[(((size_t)bb * Hdim + hh) * Ddim + d) * Ndim + ntok] = (half_t)acc[mt][nt][r];
        }
      } else {                  // Q, K: [B,H,N,D]
        half_t* dst = sel ? Kb : Qb;
        #pragma unroll
        for (int r = 0; r < 8; ++r) {
          int m = m0 + r;
          int bb = m >> 11, ntok = m & (Ndim - 1);
          dst[(((size_t)bb * Hdim + hh) * Ndim + ntok) * Ddim + d] = (half_t)acc[mt][nt][r];
        }
      }
    }
  }
}

// ---------------------------------------------------------------------------
// Flash attention: block = 256 q-rows of one (b,h); wave = 32 q-rows (2 tiles).
// K [N,D] and V^T [D,N] chunks both staged with async B128 copies,
// double-buffered so chunk c+1 streams under chunk c's 32 WMMAs.
// ---------------------------------------------------------------------------
__global__ __launch_bounds__(256) void attn_kernel(
    const half_t* __restrict__ Qb, const half_t* __restrict__ Kb,
    const half_t* __restrict__ Vb, half_t* __restrict__ attn) {
  __shared__ u32 sK[2][64][36];     // 32 data u32 + 4 pad (rows 144B)
  __shared__ u32 sV[2][64][36];     // V^T: row = d, halves along keys
  __shared__ u32 sP[8][16][36];     // per-wave P tile (16 x 64 halves)

  const int tid  = threadIdx.x;
  const int wave = tid >> 5;
  const int lane = tid & 31;
  const int hi   = lane >> 4;
  const int l15  = lane & 15;
  const int kb4  = hi * 4;
  const int kb8  = hi * 8;

  const int bh    = blockIdx.y;       // 0..47
  const int qBase = blockIdx.x * 256;
  const int bq = bh / Hdim, h = bh % Hdim;

  auto stage = [&](int kc, int buf) {
    const u32* Ku = (const u32*)Kb + ((size_t)bh * Ndim + kc) * 32;
    const u32* Vt = (const u32*)Vb + (size_t)bh * Ddim * (Ndim / 2) + (kc >> 1);
    #pragma unroll
    for (int t = 0; t < 2; ++t) {
      int id = tid + t * 256;
      int key = id >> 3, jj = (id & 7) * 4;
      async_b128((unsigned)(size_t)&sK[buf][key][jj], Ku + (size_t)key * 32 + jj);
    }
    #pragma unroll
    for (int t = 0; t < 2; ++t) {
      int id = tid + t * 256;
      int d = id >> 3, jj = (id & 7) * 4;
      async_b128((unsigned)(size_t)&sV[buf][d][jj], Vt + (size_t)d * (Ndim / 2) + jj);
    }
  };

  // Q fragments: 2 row-tiles x 2 k-steps
  FragH qa[2][2];
  #pragma unroll
  for (int qt = 0; qt < 2; ++qt) {
    const size_t qrow = (size_t)bh * Ndim + qBase + wave * 32 + qt * 16 + l15;
    const u32* qrp = (const u32*)Qb + qrow * 32;
    #pragma unroll
    for (int kk = 0; kk < 2; ++kk)
      #pragma unroll
      for (int i = 0; i < 4; ++i) {
        qa[qt][kk].u[i]     = qrp[kk * 16 + kb4 + i];
        qa[qt][kk].u[4 + i] = qrp[kk * 16 + 8 + kb4 + i];
      }
  }

  v8f acc[2][4];
  float m_run[2][8], l_run[2][8];
  #pragma unroll
  for (int qt = 0; qt < 2; ++qt) {
    #pragma unroll
    for (int ot = 0; ot < 4; ++ot) acc[qt][ot] = vzero();
    #pragma unroll
    for (int r = 0; r < 8; ++r) { m_run[qt][r] = -1e30f; l_run[qt][r] = 0.f; }
  }

  half_t* sPh = (half_t*)&sP[wave][0][0];   // row stride 72 halves
  const u32* sPw = &sP[wave][0][0];

  stage(0, 0);
  const int CHUNKS = Ndim / 64;   // 32
  for (int c = 0; c < CHUNKS; ++c) {
    wait_async0();
    __syncthreads();
    if (c + 1 < CHUNKS) stage((c + 1) * 64, (c + 1) & 1);
    const int buf = c & 1;

    // S = Q K^T : 2 x (16 q-rows) x 64 keys  (16 WMMAs, bk reused across qt)
    v8f st[2][4];
    #pragma unroll
    for (int t = 0; t < 4; ++t) {
      FragH bk[2];
      #pragma unroll
      for (int kk = 0; kk < 2; ++kk) {
        const u32* p = &sK[buf][t * 16 + l15][kk * 16];
        #pragma unroll
        for (int i = 0; i < 4; ++i) { bk[kk].u[i] = p[kb4 + i]; bk[kk].u[4 + i] = p[8 + kb4 + i]; }
      }
      #pragma unroll
      for (int qt = 0; qt < 2; ++qt) {
        v8f s = vzero();
        #pragma unroll
        for (int kk = 0; kk < 2; ++kk)
          s = __builtin_amdgcn_wmma_f32_16x16x32_f16(false, qa[qt][kk].h, false, bk[kk].h,
                                                     (short)0, s, false, false);
        st[qt][t] = s;
      }
    }

    // online softmax (row-wise across 16-lane halves)
    #pragma unroll
    for (int qt = 0; qt < 2; ++qt)
      #pragma unroll
      for (int r = 0; r < 8; ++r) {
        float mx = -1e30f;
        #pragma unroll
        for (int t = 0; t < 4; ++t) { st[qt][t][r] *= 0.125f; mx = fmaxf(mx, st[qt][t][r]); }
        #pragma unroll
        for (int off = 1; off < 16; off <<= 1) mx = fmaxf(mx, __shfl_xor(mx, off, 32));
        float mnew  = fmaxf(m_run[qt][r], mx);
        float alpha = __expf(m_run[qt][r] - mnew);
        m_run[qt][r] = mnew;
        float rs = 0.f;
        #pragma unroll
        for (int t = 0; t < 4; ++t) {
          float p = __expf(st[qt][t][r] - mnew); st[qt][t][r] = p; rs += p;
        }
        #pragma unroll
        for (int off = 1; off < 16; off <<= 1) rs += __shfl_xor(rs, off, 32);
        l_run[qt][r] = l_run[qt][r] * alpha + rs;
        #pragma unroll
        for (int ot = 0; ot < 4; ++ot) acc[qt][ot][r] *= alpha;
      }

    // per q-tile: stage P (wave-private, LDS is in-order per wave), then O += P V
    #pragma unroll
    for (int qt = 0; qt < 2; ++qt) {
      #pragma unroll
      for (int t = 0; t < 4; ++t)
        #pragma unroll
        for (int r = 0; r < 8; ++r)
          sPh[(r + 8 * hi) * 72 + t * 16 + l15] = (half_t)st[qt][t][r];

      #pragma unroll
      for (int kk2 = 0; kk2 < 2; ++kk2) {
        FragH pa;
        #pragma unroll
        for (int i = 0; i < 4; ++i) {
          pa.u[i]     = sPw[l15 * 36 + kk2 * 16 + kb4 + i];
          pa.u[4 + i] = sPw[l15 * 36 + kk2 * 16 + 8 + kb4 + i];
        }
        #pragma unroll
        for (int ot = 0; ot < 4; ++ot) {
          FragH bv;
          const u32* p = &sV[buf][ot * 16 + l15][kk2 * 16];
          #pragma unroll
          for (int i = 0; i < 4; ++i) { bv.u[i] = p[kb4 + i]; bv.u[4 + i] = p[8 + kb4 + i]; }
          acc[qt][ot] = __builtin_amdgcn_wmma_f32_16x16x32_f16(false, pa.h, false, bv.h,
                                                               (short)0, acc[qt][ot], false, false);
        }
      }
    }
  }

  // normalize + write f16 [B,N,C]
  #pragma unroll
  for (int qt = 0; qt < 2; ++qt)
    #pragma unroll
    for (int r = 0; r < 8; ++r) {
      float inv = 1.0f / l_run[qt][r];
      int ntok = qBase + wave * 32 + qt * 16 + r + 8 * hi;
      size_t rowoff = ((size_t)bq * Ndim + ntok) * Cdim + h * Ddim;
      #pragma unroll
      for (int ot = 0; ot < 4; ++ot)
        attn[rowoff + ot * 16 + l15] = (half_t)(acc[qt][ot][r] * inv);
    }
}

// ---------------------------------------------------------------------------
// Output GEMM: attn[8192,768] @ woutT[768,768] + b_out + residual -> f32 out
// Same geometry as QKV GEMM (double-buffered async staging).
// ---------------------------------------------------------------------------
__global__ __launch_bounds__(256) void gemm_out_kernel(
    const half_t* __restrict__ A, const half_t* __restrict__ BT,
    const float* __restrict__ bias, const float* __restrict__ resid,
    float* __restrict__ out) {
  __shared__ u32 sA[2][256][20];
  __shared__ u32 sB[2][128][20];

  const int tid = threadIdx.x;
  const int wave = tid >> 5, lane = tid & 31;
  const int hi = lane >> 4, l15 = lane & 15;
  const int kb4 = hi * 4, kb8 = hi * 8;
  const int wm = wave >> 1, wn = wave & 1;
  const int mBase = blockIdx.x * 256;
  const int nBase = blockIdx.y * 128;
  const u32* Au = (const u32*)A;
  const u32* Bu = (const u32*)BT;

  auto stage = [&](int k0, int buf) {
    #pragma unroll
    for (int t = 0; t < 4; ++t) {
      int id = tid + t * 256;
      int r = id >> 2, c4 = (id & 3) * 4;
      async_b128((unsigned)(size_t)&sA[buf][r][c4],
                 Au + (size_t)(mBase + r) * (Cdim / 2) + (k0 >> 1) + c4);
    }
    #pragma unroll
    for (int t = 0; t < 2; ++t) {
      int id = tid + t * 256;
      int r = id >> 2, c4 = (id & 3) * 4;
      async_b128((unsigned)(size_t)&sB[buf][r][c4],
                 Bu + (size_t)(nBase + r) * (Cdim / 2) + (k0 >> 1) + c4);
    }
  };

  v8f acc[4][4];
  #pragma unroll
  for (int mt = 0; mt < 4; ++mt)
    #pragma unroll
    for (int nt = 0; nt < 4; ++nt) acc[mt][nt] = vzero();

  stage(0, 0);
  const int KSTEPS = Cdim / 32;
  for (int ks = 0; ks < KSTEPS; ++ks) {
    wait_async0();
    __syncthreads();
    if (ks + 1 < KSTEPS) stage((ks + 1) * 32, (ks + 1) & 1);
    const int buf = ks & 1;

    FragH af[4], bf[4];
    #pragma unroll
    for (int t = 0; t < 4; ++t) {
      const u32* p = &sA[buf][wm * 64 + t * 16 + l15][0];
      #pragma unroll
      for (int i = 0; i < 4; ++i) { af[t].u[i] = p[kb4 + i]; af[t].u[4 + i] = p[8 + kb4 + i]; }
    }
    #pragma unroll
    for (int t = 0; t < 4; ++t) {
      const u32* p = &sB[buf][wn * 64 + t * 16 + l15][0];
      #pragma unroll
      for (int i = 0; i < 4; ++i) { bf[t].u[i] = p[kb4 + i]; bf[t].u[4 + i] = p[8 + kb4 + i]; }
    }
    #pragma unroll
    for (int mt = 0; mt < 4; ++mt)
      #pragma unroll
      for (int nt = 0; nt < 4; ++nt)
        acc[mt][nt] = __builtin_amdgcn_wmma_f32_16x16x32_f16(
            false, af[mt].h, false, bf[nt].h, (short)0, acc[mt][nt], false, false);
  }

  #pragma unroll
  for (int mt = 0; mt < 4; ++mt) {
    #pragma unroll
    for (int nt = 0; nt < 4; ++nt) {
      int col = nBase + wn * 64 + nt * 16 + l15;
      float bcol = bias[col];
      int m0 = mBase + wm * 64 + mt * 16 + hi * 8;
      #pragma unroll
      for (int r = 0; r < 8; ++r) {
        int m = m0 + r;
        size_t o = (size_t)m * Cdim + col;
        out[o] = acc[mt][nt][r] + bcol + resid[o];
      }
    }
  }
}

// ---------------------------------------------------------------------------
// Launch
// ---------------------------------------------------------------------------
extern "C" void kernel_launch(void* const* d_in, const int* in_sizes, int n_in,
                              void* d_out, int out_size, void* d_ws, size_t ws_size,
                              hipStream_t stream) {
  const float* x    = (const float*)d_in[0];
  const float* g    = (const float*)d_in[1];
  const float* be   = (const float*)d_in[2];
  const float* wqkv = (const float*)d_in[3];
  const float* wout = (const float*)d_in[4];
  const float* bout = (const float*)d_in[5];
  float* out = (float*)d_out;

  // workspace layout (bytes), total ~67.6 MB
  constexpr size_t XN_SZ    = (size_t)Mdim * Cdim * 2;
  constexpr size_t WQKVT_SZ = (size_t)Cdim * 3 * Cdim * 2;
  constexpr size_t WOUTT_SZ = (size_t)Cdim * Cdim * 2;
  char* ws = (char*)d_ws;
  half_t* xn    = (half_t*)(ws);
  half_t* wqkvT = (half_t*)(ws + XN_SZ);
  half_t* woutT = (half_t*)(ws + XN_SZ + WQKVT_SZ);
  half_t* Qb    = (half_t*)(ws + XN_SZ + WQKVT_SZ + WOUTT_SZ);
  half_t* Kb    = (half_t*)(ws + XN_SZ + WQKVT_SZ + WOUTT_SZ + XN_SZ);
  half_t* Vb    = (half_t*)(ws + XN_SZ + WQKVT_SZ + WOUTT_SZ + 2 * XN_SZ);
  half_t* attn  = (half_t*)(ws + XN_SZ + WQKVT_SZ + WOUTT_SZ + 3 * XN_SZ);

  transpose_to_h<<<(Cdim * 3 * Cdim + 255) / 256, 256, 0, stream>>>(wqkv, wqkvT, Cdim, 3 * Cdim);
  transpose_to_h<<<(Cdim * Cdim + 255) / 256, 256, 0, stream>>>(wout, woutT, Cdim, Cdim);
  ln_kernel<<<Mdim, 256, 0, stream>>>(x, g, be, xn);
  gemm_qkv_kernel<<<dim3(Mdim / 256, (3 * Cdim) / 128), 256, 0, stream>>>(xn, wqkvT, Qb, Kb, Vb);
  attn_kernel<<<dim3(Ndim / 256, Bdim * Hdim), 256, 0, stream>>>(Qb, Kb, Vb, attn);
  gemm_out_kernel<<<dim3(Mdim / 256, Cdim / 128), 256, 0, stream>>>(attn, woutT, bout, x, out);
}
